// CGRNN_FB_256_66228395704671
// MI455X (gfx1250) — matrix operand order repaired and checked by reading, compiled, not run on hardware
//
#include <hip/hip_runtime.h>
#include <math.h>

typedef __attribute__((ext_vector_type(16))) _Float16 v16h;
typedef __attribute__((ext_vector_type(8)))  _Float16 v8h;
typedef __attribute__((ext_vector_type(8)))  float    v8f;

#define BSZ 32
#define TLEN 1000

// ---------------------------------------------------------------------------
// f32 -> f16 conversion (weights)
// ---------------------------------------------------------------------------
__global__ __launch_bounds__(256) void k_cvt_f16(
    const float* __restrict__ src, _Float16* __restrict__ dst, int n) {
  int i = blockIdx.x * 256 + threadIdx.x;
  if (i < n) dst[i] = (_Float16)src[i];
}

// ---------------------------------------------------------------------------
// Generic strided conv (k=3, stride 2 on F) + BatchNorm + LeakyReLU(0.3)
// x: (B, Cin, T, Fin)  w: (Cout, Cin, 1, 3)  y: (B, Cout, T, Fout)
// If gin_h != nullptr (conv3 only): also write f16 gin[b,t, o*Fout+fo]
// ---------------------------------------------------------------------------
__global__ __launch_bounds__(256) void k_conv_bn_lrelu(
    const float* __restrict__ xin, const float* __restrict__ w,
    const float* __restrict__ cb, const float* __restrict__ bg,
    const float* __restrict__ bbeta, const float* __restrict__ bm,
    const float* __restrict__ bv, float* __restrict__ y,
    _Float16* __restrict__ gin_h,
    int Cin, int Cout, int Fin, int Fout, int pad, int total) {
  int idx = blockIdx.x * 256 + threadIdx.x;
  if (idx >= total) return;
  int fo = idx % Fout; int tmp = idx / Fout;
  int t = tmp % TLEN; tmp /= TLEN;
  int o = tmp % Cout; int b = tmp / Cout;
  float acc = cb[o];
  int base = fo * 2 - pad;
  for (int c = 0; c < Cin; ++c) {
    const float* xr_ = &xin[((b * Cin + c) * TLEN + t) * Fin];
    const float* wr  = &w[(o * Cin + c) * 3];
#pragma unroll
    for (int k = 0; k < 3; ++k) {
      int fi = base + k;
      if (fi >= 0 && fi < Fin) acc += xr_[fi] * wr[k];
    }
  }
  float s  = bg[o] * rsqrtf(bv[o] + 1e-5f);
  float yv = (acc - bm[o]) * s + bbeta[o];
  float outv = yv >= 0.f ? yv : 0.3f * yv;
  y[idx] = outv;
  if (gin_h)  // conv3: gin[b,t, c3*16+f3] = x3 transposed, as f16 for WMMA
    gin_h[(b * TLEN + t) * 512 + o * Fout + fo] = (_Float16)outv;
}

// ---------------------------------------------------------------------------
// GRU input projection: xp[b,t,g,o] = sum_i gin[b,t,g*64+i] * wih[g,o,i] + bih
// WMMA f32<-f16 16x16x32, fragments loaded as contiguous b128 chunks.
// One wave handles 16 rows x 48 cols (3 N-tiles, A reused).
// Waves: G(8) * MT(2000) * 4 = 64000 -> 16000 blocks of 128.
// ---------------------------------------------------------------------------
__global__ __launch_bounds__(128) void k_xp_gemm(
    const _Float16* __restrict__ gin_h, const _Float16* __restrict__ wih_h,
    const float* __restrict__ bih, float* __restrict__ xp) {
  const int lane = threadIdx.x & 31;
  const int wave = threadIdx.x >> 5;
  int wv  = blockIdx.x * 4 + wave;
  int g   = wv / 8000;
  int rem = wv % 8000;
  int m0  = (rem >> 2) * 16;
  int nb0 = (rem & 3) * 48;

  const int am  = lane & 15;
  const int akb = (lane & 16) ? 16 : 0;
  const int bn  = lane & 15;
  const int bkb = (lane & 16) ? 16 : 0;
  const int dmb = (lane & 16) ? 8 : 0;
  const int dn  = lane & 15;

  const _Float16* arow = &gin_h[(m0 + am) * 512 + g * 64];

  v8f acc0 = {}, acc1 = {}, acc2 = {};
#pragma unroll
  for (int k0 = 0; k0 < 64; k0 += 32) {
    v8h a0 = *(const v8h*)&arow[k0 + akb];
    v8h a1 = *(const v8h*)&arow[k0 + 8 + akb];
    v16h a;
#pragma unroll
    for (int e = 0; e < 8; ++e) { a[e] = a0[e]; a[8 + e] = a1[e]; }
    v16h bf0 = *(const v16h*)&wih_h[(g * 192 + nb0 +      bn) * 64 + k0 + bkb];
    v16h bf1 = *(const v16h*)&wih_h[(g * 192 + nb0 + 16 + bn) * 64 + k0 + bkb];
    v16h bf2 = *(const v16h*)&wih_h[(g * 192 + nb0 + 32 + bn) * 64 + k0 + bkb];
    acc0 = __builtin_amdgcn_wmma_f32_16x16x32_f16(false, a, false, bf0, (short)0, acc0, false, false);
    acc1 = __builtin_amdgcn_wmma_f32_16x16x32_f16(false, a, false, bf1, (short)0, acc1, false, false);
    acc2 = __builtin_amdgcn_wmma_f32_16x16x32_f16(false, a, false, bf2, (short)0, acc2, false, false);
  }
#pragma unroll
  for (int j = 0; j < 3; ++j) {
    int n0 = nb0 + j * 16;
    v8f acc = (j == 0) ? acc0 : (j == 1) ? acc1 : acc2;
    float bias = bih[g * 192 + n0 + dn];
#pragma unroll
    for (int r = 0; r < 8; ++r) {
      int m = m0 + dmb + r;
      int b2 = m / TLEN, t2 = m % TLEN;
      xp[((b2 * TLEN + t2) * 8 + g) * 192 + n0 + dn] = acc[r] + bias;
    }
  }
}

// ---------------------------------------------------------------------------
// Persistent grouped-GRU scan. Groups independent: 8 blocks, 256 threads.
// whh (f16) + h state + hp live in LDS; 24 WMMA tiles/step, b128 LDS loads.
// ---------------------------------------------------------------------------
__global__ __launch_bounds__(256) void k_gru_scan(
    const float* __restrict__ xp, const float* __restrict__ whh,
    const float* __restrict__ bhh, float* __restrict__ gout) {
  const int g = blockIdx.x;
  const int tid = threadIdx.x;
  const int lane = tid & 31;
  const int wave = tid >> 5;

  __shared__ __align__(32) _Float16 whh_s[192 * 64];   // [o][h]
  __shared__ float    bhh_s[192];
  __shared__ float    h_s[32 * 64];
  __shared__ __align__(32) _Float16 h_h[32 * 64];
  __shared__ float    hp_s[32 * 192];

  for (int i = tid; i < 192 * 64; i += 256) whh_s[i] = (_Float16)whh[g * 192 * 64 + i];
  for (int i = tid; i < 192; i += 256) bhh_s[i] = bhh[g * 192 + i];
  for (int i = tid; i < 2048; i += 256) { h_s[i] = 0.f; h_h[i] = (_Float16)0.f; }
  __syncthreads();

  const int am  = lane & 15;
  const int akb = (lane & 16) ? 16 : 0;
  const int bn  = lane & 15;
  const int bkb = (lane & 16) ? 16 : 0;
  const int dmb = (lane & 16) ? 8 : 0;
  const int dn  = lane & 15;

  for (int t = 0; t < TLEN; ++t) {
#pragma unroll
    for (int tt = 0; tt < 3; ++tt) {          // 8 waves * 3 tiles = 24 tiles
      int tile = wave * 3 + tt;
      int m0 = (tile / 12) * 16, n0 = (tile % 12) * 16;
      const _Float16* hrow = &h_h[(m0 + am) * 64];
      const _Float16* wrow = &whh_s[(n0 + bn) * 64];
      v8f acc = {};
#pragma unroll
      for (int k0 = 0; k0 < 64; k0 += 32) {
        v8h a0 = *(const v8h*)&hrow[k0 + akb];
        v8h a1 = *(const v8h*)&hrow[k0 + 8 + akb];
        v16h a;
#pragma unroll
        for (int e = 0; e < 8; ++e) { a[e] = a0[e]; a[8 + e] = a1[e]; }
        v16h bf = *(const v16h*)&wrow[k0 + bkb];
        acc = __builtin_amdgcn_wmma_f32_16x16x32_f16(false, a, false, bf,
                                                     (short)0, acc, false, false);
      }
      float bias = bhh_s[n0 + dn];
#pragma unroll
      for (int r2 = 0; r2 < 8; ++r2)
        hp_s[(m0 + dmb + r2) * 192 + n0 + dn] = acc[r2] + bias;
    }
    __syncthreads();

    for (int e = tid; e < 2048; e += 256) {
      int b = e >> 6, hh = e & 63;
      const float* xpt = &xp[((b * TLEN + t) * 8 + g) * 192];
      if (t + 1 < TLEN) {  // hide next step's only HBM traffic
        const float* xpn = &xp[((b * TLEN + (t + 1)) * 8 + g) * 192];
        __builtin_prefetch(xpn + hh, 0, 1);
        __builtin_prefetch(xpn + 128 + hh, 0, 1);
      }
      float xr = xpt[hh], xz = xpt[64 + hh], xn = xpt[128 + hh];
      float hr = hp_s[b * 192 + hh];
      float hz = hp_s[b * 192 + 64 + hh];
      float hn = hp_s[b * 192 + 128 + hh];
      float r = 1.f / (1.f + __expf(-(xr + hr)));
      float z = 1.f / (1.f + __expf(-(xz + hz)));
      float n = tanhf(xn + r * hn);
      float hnew = (1.f - z) * n + z * h_s[e];
      h_s[e] = hnew;
      h_h[e] = (_Float16)hnew;
      gout[(b * TLEN + t) * 512 + g * 64 + hh] = hnew;
    }
    __syncthreads();
  }
}

// ---------------------------------------------------------------------------
// convT1 + BN + lrelu fused with concat -> r1cat (B,32,T,32)
// ---------------------------------------------------------------------------
__global__ __launch_bounds__(256) void k_convT1(
    const float* __restrict__ gout, const float* __restrict__ x3,
    const float* __restrict__ x2, const float* __restrict__ w,
    const float* __restrict__ cb, const float* __restrict__ g_,
    const float* __restrict__ be_, const float* __restrict__ m_,
    const float* __restrict__ v_, float* __restrict__ r1cat) {
  int idx = blockIdx.x * 256 + threadIdx.x;
  const int total = BSZ * 32 * TLEN * 32;
  if (idx >= total) return;
  int fo = idx & 31; int tmp = idx >> 5;
  int t = tmp % TLEN; tmp /= TLEN;
  int c = tmp & 31; int b = tmp >> 5;
  float out;
  if (c < 16) {
    int o = c;
    float acc = cb[o];
#pragma unroll
    for (int k = 0; k < 3; ++k) {
      int p = fo - 2 + k;
      if (p >= 0 && (p & 1) == 0) {
        int fi = p >> 1;
        if (fi < 16) {
          int kk = 2 - k;
          const float* gp = &gout[(b * TLEN + t) * 512 + fi];
          for (int i = 0; i < 32; ++i)
            acc += gp[i * 16] * w[(i * 16 + o) * 3 + kk];
          for (int i = 0; i < 32; ++i)
            acc += x3[((b * 32 + i) * TLEN + t) * 16 + fi] * w[((i + 32) * 16 + o) * 3 + kk];
        }
      }
    }
    float s  = g_[o] * rsqrtf(v_[o] + 1e-5f);
    float yv = (acc - m_[o]) * s + be_[o];
    out = yv >= 0.f ? yv : 0.3f * yv;
  } else {
    out = x2[((b * 16 + (c - 16)) * TLEN + t) * 33 + fo];
  }
  r1cat[idx] = out;
}

// ---------------------------------------------------------------------------
// convT2 + BN + lrelu fused with concat -> r2cat (B,16,T,64)
// ---------------------------------------------------------------------------
__global__ __launch_bounds__(256) void k_convT2(
    const float* __restrict__ r1cat, const float* __restrict__ x1,
    const float* __restrict__ w, const float* __restrict__ cb,
    const float* __restrict__ g_, const float* __restrict__ be_,
    const float* __restrict__ m_, const float* __restrict__ v_,
    float* __restrict__ r2cat) {
  int idx = blockIdx.x * 256 + threadIdx.x;
  const int total = BSZ * 16 * TLEN * 64;
  if (idx >= total) return;
  int fo = idx & 63; int tmp = idx >> 6;
  int t = tmp % TLEN; tmp /= TLEN;
  int c = tmp & 15; int b = tmp >> 4;
  float out;
  if (c < 8) {
    int o = c;
    float acc = cb[o];
#pragma unroll
    for (int k = 0; k < 3; ++k) {
      int p = fo - 2 + k;
      if (p >= 0 && (p & 1) == 0) {
        int fi = p >> 1;
        if (fi < 32) {
          int kk = 2 - k;
          for (int i = 0; i < 32; ++i)
            acc += r1cat[((b * 32 + i) * TLEN + t) * 32 + fi] * w[(i * 8 + o) * 3 + kk];
        }
      }
    }
    float s  = g_[o] * rsqrtf(v_[o] + 1e-5f);
    float yv = (acc - m_[o]) * s + be_[o];
    out = yv >= 0.f ? yv : 0.3f * yv;
  } else {
    out = x1[((b * 8 + (c - 8)) * TLEN + t) * 65 + fo];
  }
  r2cat[idx] = out;
}

// ---------------------------------------------------------------------------
// convT3 (2 output channels) fused with the complex mask multiply -> d_out
// ---------------------------------------------------------------------------
__global__ __launch_bounds__(256) void k_convT3_mask(
    const float* __restrict__ r2cat, const float* __restrict__ x,
    const float* __restrict__ w, const float* __restrict__ cb,
    float* __restrict__ out) {
  int idx = blockIdx.x * 256 + threadIdx.x;
  const int total = BSZ * TLEN * 129;
  if (idx >= total) return;
  int f = idx % 129; int tmp = idx / 129;
  int t = tmp % TLEN; int b = tmp / TLEN;
  float mr = cb[0], mi = cb[1];
#pragma unroll
  for (int k = 0; k < 3; ++k) {
    int p = f - 2 + k;
    if (p >= 0 && (p & 1) == 0) {
      int fi = p >> 1;
      if (fi < 64) {
        int kk = 2 - k;
        for (int i = 0; i < 16; ++i) {
          float xv = r2cat[((b * 16 + i) * TLEN + t) * 64 + fi];
          mr += xv * w[(i * 2 + 0) * 3 + kk];
          mi += xv * w[(i * 2 + 1) * 3 + kk];
        }
      }
    }
  }
  float nr = x[((b * 2 + 0) * TLEN + t) * 129 + f];
  float ni = x[((b * 2 + 1) * TLEN + t) * 129 + f];
  out[(b * TLEN + t) * 129 + f]           = nr * mr - ni * mi;
  out[4128000 + (b * TLEN + t) * 129 + f] = nr * mi + ni * mr;
}

// ---------------------------------------------------------------------------
extern "C" void kernel_launch(void* const* d_in, const int* in_sizes, int n_in,
                              void* d_out, int out_size, void* d_ws, size_t ws_size,
                              hipStream_t stream) {
  (void)in_sizes; (void)n_in; (void)out_size; (void)ws_size;
  const float* x   = (const float*)d_in[0];
  const float* w1  = (const float*)d_in[1];  const float* b1  = (const float*)d_in[2];
  const float* g1  = (const float*)d_in[3];  const float* be1 = (const float*)d_in[4];
  const float* m1  = (const float*)d_in[5];  const float* v1  = (const float*)d_in[6];
  const float* w2  = (const float*)d_in[7];  const float* b2  = (const float*)d_in[8];
  const float* g2  = (const float*)d_in[9];  const float* be2 = (const float*)d_in[10];
  const float* m2  = (const float*)d_in[11]; const float* v2  = (const float*)d_in[12];
  const float* w3  = (const float*)d_in[13]; const float* b3  = (const float*)d_in[14];
  const float* g3  = (const float*)d_in[15]; const float* be3 = (const float*)d_in[16];
  const float* m3  = (const float*)d_in[17]; const float* v3  = (const float*)d_in[18];
  const float* wih = (const float*)d_in[19]; const float* whh = (const float*)d_in[20];
  const float* bih = (const float*)d_in[21]; const float* bhh = (const float*)d_in[22];
  const float* wT1 = (const float*)d_in[23]; const float* bT1 = (const float*)d_in[24];
  const float* gT1 = (const float*)d_in[25]; const float* beT1= (const float*)d_in[26];
  const float* mT1 = (const float*)d_in[27]; const float* vT1 = (const float*)d_in[28];
  const float* wT2 = (const float*)d_in[29]; const float* bT2 = (const float*)d_in[30];
  const float* gT2 = (const float*)d_in[31]; const float* beT2= (const float*)d_in[32];
  const float* mT2 = (const float*)d_in[33]; const float* vT2 = (const float*)d_in[34];
  const float* wT3 = (const float*)d_in[35]; const float* bT3 = (const float*)d_in[36];

  float* ws    = (float*)d_ws;
  float* x1    = ws;                        // 16,640,000  (B,8,T,65)
  float* x2    = ws + 16640000;             // 16,896,000  (B,16,T,33)
  float* x3    = ws + 33536000;             // 16,384,000  (B,32,T,16)
  float* gout  = ws + 49920000;             // 16,384,000  (B,T,512)
  float* xp    = ws + 66304000;             // 49,152,000  (B,T,G,192)
  _Float16* gin_h = (_Float16*)(ws + 115456000);  // 16,384,000 halves (B,T,512)
  _Float16* wih_h = (_Float16*)(ws + 123648000);  //     98,304 halves (G,192,64)
  float* r1cat = xp;                        // alias: xp dead after scan
  float* r2cat = x2;                        // alias: x2+x3 dead after convT1
  float* outp  = (float*)d_out;

  // weight conversion + encoder convs
  k_cvt_f16<<<(98304 + 255) / 256, 256, 0, stream>>>(wih, wih_h, 98304);
  k_conv_bn_lrelu<<<(16640000 + 255) / 256, 256, 0, stream>>>(
      x,  w1, b1, g1, be1, m1, v1, x1, nullptr, 2,  8, 129, 65, 1, 16640000);
  k_conv_bn_lrelu<<<(16896000 + 255) / 256, 256, 0, stream>>>(
      x1, w2, b2, g2, be2, m2, v2, x2, nullptr, 8, 16,  65, 33, 1, 16896000);
  k_conv_bn_lrelu<<<(16384000 + 255) / 256, 256, 0, stream>>>(
      x2, w3, b3, g3, be3, m3, v3, x3, gin_h, 16, 32, 33, 16, 0, 16384000);

  // GRU input projection (WMMA, b128 fragment loads) + persistent scan
  k_xp_gemm<<<16000, 128, 0, stream>>>(gin_h, wih_h, bih, xp);
  k_gru_scan<<<8, 256, 0, stream>>>(xp, whh, bhh, gout);

  // decoder
  k_convT1<<<(32768000 + 255) / 256, 256, 0, stream>>>(
      gout, x3, x2, wT1, bT1, gT1, beT1, mT1, vT1, r1cat);
  k_convT2<<<(32768000 + 255) / 256, 256, 0, stream>>>(
      r1cat, x1, wT2, bT2, gT2, beT2, mT2, vT2, r2cat);
  k_convT3_mask<<<(4128000 + 255) / 256, 256, 0, stream>>>(
      r2cat, x, wT3, bT3, outp);
}
